// Rollout_8529805049990
// MI455X (gfx1250) — matrix operand order repaired and checked by reading, not compiled, not run on hardware
//
#include <hip/hip_runtime.h>
#include <hip/hip_bf16.h>

typedef __attribute__((ext_vector_type(16))) __bf16 v16bf;
typedef __attribute__((ext_vector_type(8)))  float  v8f;

#define B_SZ   64
#define T_LEN  128
#define E_DIM  256
#define H_DIM  1024
#define G4     4096            // 4*H
#define KCAT   1536            // 2E + H
#define V_SZ   32000
#define NT_OUT (V_SZ / 16)     // 2000 N-tiles for the output projection

union FragU { v16bf v; uint4 q[2]; };

__device__ __forceinline__ v8f wmma_bf16(FragU a, FragU b, v8f c) {
  // D = A(16x32 bf16) * B(32x16 bf16) + C(16x16 f32)
  return __builtin_amdgcn_wmma_f32_16x16x32_bf16(false, a.v, false, b.v,
                                                 (short)0, c, false, false);
}

__device__ __forceinline__ float gumbel_noise(int t, int b, int v) {
  unsigned h = (unsigned)t * 0x9E3779B1u + (unsigned)b * 0x85EBCA77u +
               (unsigned)v * 0xC2B2AE3Du + 0x27220A95u;
  h ^= h >> 16; h *= 0x7FEB352Du; h ^= h >> 15; h *= 0x846CA68Bu; h ^= h >> 16;
  float u = ((float)h + 0.5f) * 2.3283064e-10f;   // (0,1)
  return -__logf(-__logf(u));
}

__device__ __forceinline__ unsigned long long pack_score(float s, int v) {
  unsigned u = __float_as_uint(s);
  u = (u & 0x80000000u) ? ~u : (u | 0x80000000u); // monotone float->uint
  return ((unsigned long long)u << 32) | (unsigned)v;
}

// ---------------- one-time setup (per launch) ----------------

// W_cat[n][k] = k<512 ? W_ih[n][k] : W_hh[n][k-512], converted to bf16
__global__ void k_cvt_wcat(const float* __restrict__ W_ih,
                           const float* __restrict__ W_hh,
                           __hip_bfloat16* __restrict__ Wcat) {
  int i = blockIdx.x * blockDim.x + threadIdx.x;     // < 4096*1536
  int n = i / KCAT, k = i % KCAT;
  float f = (k < 2 * E_DIM) ? W_ih[n * (2 * E_DIM) + k]
                            : W_hh[n * H_DIM + (k - 2 * E_DIM)];
  Wcat[i] = __float2bfloat16(f);
}

__global__ void k_cvt_wout(const float* __restrict__ W,
                           __hip_bfloat16* __restrict__ Wo) {
  int i = (blockIdx.x * blockDim.x + threadIdx.x) * 4;  // < 32000*1024
  float4 f = *reinterpret_cast<const float4*>(W + i);
  Wo[i + 0] = __float2bfloat16(f.x);
  Wo[i + 1] = __float2bfloat16(f.y);
  Wo[i + 2] = __float2bfloat16(f.z);
  Wo[i + 3] = __float2bfloat16(f.w);
}

__global__ void k_setup(const float* __restrict__ cls_tab,
                        const int* __restrict__ cls_lab,
                        const int* __restrict__ x,
                        const float* __restrict__ b_ih,
                        const float* __restrict__ b_hh,
                        float* __restrict__ c, __hip_bfloat16* __restrict__ xin,
                        float* __restrict__ bcat, int* __restrict__ prev) {
  int tid = blockIdx.x * blockDim.x + threadIdx.x;   // 65536 threads
  c[tid] = 0.f;                                      // c0 = 0
  int b = tid >> 10, j = tid & 1023;
  xin[b * KCAT + 2 * E_DIM + j] = __float2bfloat16(0.f);  // h0 = 0
  if (tid < B_SZ * E_DIM) {                          // class emb (constant)
    int bb = tid >> 8, jj = tid & 255;
    xin[bb * KCAT + E_DIM + jj] =
        __float2bfloat16(cls_tab[cls_lab[bb] * E_DIM + jj]);
  }
  if (tid < G4) bcat[tid] = b_ih[tid] + b_hh[tid];
  if (tid < B_SZ) prev[tid] = x[tid * T_LEN];        // prev_tok = x[:,0]
}

// ---------------- per-step kernels ----------------

__global__ void k_build_input(int t, const int* __restrict__ x,
                              const int* __restrict__ given,
                              const int* __restrict__ prev,
                              const float* __restrict__ emb,
                              __hip_bfloat16* __restrict__ xin) {
  int b = blockIdx.x, j = threadIdx.x;               // 64 x 256
  int tok = (t <= *given) ? x[b * T_LEN + t] : prev[b];
  xin[b * KCAT + j] = __float2bfloat16(emb[tok * E_DIM + j]);
}

// gates = [x_t|h] @ W_cat^T ; one wave per (m-tile, n-tile); 1024 waves
__global__ void k_gates(const __hip_bfloat16* __restrict__ xin,
                        const __hip_bfloat16* __restrict__ Wcat,
                        float* __restrict__ gates) {
  const int wave = (blockIdx.x * blockDim.x + threadIdx.x) >> 5;
  const int lane = threadIdx.x & 31;
  const int mt = wave & 3;          // 0..3  (M = 64)
  const int nt = wave >> 2;         // 0..255
  const int lhalf = lane >> 4, l15 = lane & 15;
  const __hip_bfloat16* aptr = xin + (mt * 16 + l15) * KCAT + (lhalf << 3);
  const __hip_bfloat16* bptr = Wcat + (nt * 16 + l15) * KCAT + (lhalf << 3);
  v8f acc = {};
  for (int kb = 0; kb < KCAT; kb += 32) {
    FragU a, b;
    a.q[0] = *reinterpret_cast<const uint4*>(aptr + kb);
    a.q[1] = *reinterpret_cast<const uint4*>(aptr + kb + 16);
    b.q[0] = *reinterpret_cast<const uint4*>(bptr + kb);
    b.q[1] = *reinterpret_cast<const uint4*>(bptr + kb + 16);
    __builtin_prefetch(bptr + kb + 128, 0, 1);
    acc = wmma_bf16(a, b, acc);
  }
  const int n = nt * 16 + l15;
  const int mbase = mt * 16 + (lhalf << 3);
#pragma unroll
  for (int r = 0; r < 8; ++r) gates[(mbase + r) * G4 + n] = acc[r];
}

// LSTM pointwise: c' = sig(f)*c + sig(i)*tanh(g); h = sig(o)*tanh(c')
__global__ void k_cell(const float* __restrict__ gates,
                       const float* __restrict__ bcat, float* __restrict__ c,
                       __hip_bfloat16* __restrict__ xin,
                       unsigned long long* __restrict__ best) {
  int tid = blockIdx.x * blockDim.x + threadIdx.x;   // 65536
  int b = tid >> 10, j = tid & 1023;
  const float* g = gates + b * G4;
  float ig = g[j]          + bcat[j];
  float fg = g[j + H_DIM]  + bcat[j + H_DIM];
  float gg = g[j + 2*H_DIM]+ bcat[j + 2*H_DIM];
  float og = g[j + 3*H_DIM]+ bcat[j + 3*H_DIM];
  float si = 1.f / (1.f + __expf(-ig));
  float sf = 1.f / (1.f + __expf(-fg));
  float so = 1.f / (1.f + __expf(-og));
  float cc = sf * c[tid] + si * tanhf(gg);
  c[tid] = cc;
  xin[b * KCAT + 2 * E_DIM + j] = __float2bfloat16(so * tanhf(cc));
  if (tid < B_SZ) best[tid] = 0ull;                  // reset sampler
}

// logits = h @ W_out^T ; Gumbel-argmax sampling via hierarchical max
// one wave per 16-col tile of W_out, 4 M-tiles per wave (W_out read once)
__global__ void k_logits(int t, const __hip_bfloat16* __restrict__ xin,
                         const __hip_bfloat16* __restrict__ Wo,
                         const float* __restrict__ b_out,
                         unsigned long long* __restrict__ best) {
  __shared__ unsigned long long sbest[B_SZ];
  if (threadIdx.x < B_SZ) sbest[threadIdx.x] = 0ull;
  __syncthreads();

  const int wave = (blockIdx.x * blockDim.x + threadIdx.x) >> 5;  // 0..1999
  const int lane = threadIdx.x & 31;
  const int lhalf = lane >> 4, l15 = lane & 15;
  const __hip_bfloat16* bptr = Wo + (wave * 16 + l15) * H_DIM + (lhalf << 3);
  const __hip_bfloat16* aptr = xin + l15 * KCAT + 2 * E_DIM + (lhalf << 3);

  v8f acc[4] = {};
  for (int kb = 0; kb < H_DIM; kb += 32) {
    FragU bf;
    bf.q[0] = *reinterpret_cast<const uint4*>(bptr + kb);
    bf.q[1] = *reinterpret_cast<const uint4*>(bptr + kb + 16);
    __builtin_prefetch(bptr + kb + 128, 0, 1);       // global_prefetch_b8
#pragma unroll
    for (int mtl = 0; mtl < 4; ++mtl) {
      FragU af;
      const __hip_bfloat16* ap = aptr + mtl * (16 * KCAT) + kb;
      af.q[0] = *reinterpret_cast<const uint4*>(ap);
      af.q[1] = *reinterpret_cast<const uint4*>(ap + 16);
      acc[mtl] = wmma_bf16(af, bf, acc[mtl]);
    }
  }

  const int v  = wave * 16 + l15;
  const float bo = b_out[v];
  const int mh = lhalf << 3;
#pragma unroll
  for (int mtl = 0; mtl < 4; ++mtl) {
#pragma unroll
    for (int r = 0; r < 8; ++r) {
      const int brow = mtl * 16 + mh + r;
      unsigned long long p =
          pack_score(acc[mtl][r] + bo + gumbel_noise(t, brow, v), v);
#pragma unroll
      for (int m = 1; m < 16; m <<= 1) {             // 16-lane max reduce
        unsigned long long q = __shfl_xor(p, m, 32);
        p = (q > p) ? q : p;
      }
      if (l15 == 0) atomicMax(&sbest[brow], p);      // ds_max_u64
    }
  }
  __syncthreads();
  if (threadIdx.x < B_SZ)
    atomicMax(&best[threadIdx.x], sbest[threadIdx.x]); // global_atomic_max_u64
}

__global__ void k_finalize(int t, const int* __restrict__ x,
                           const int* __restrict__ given,
                           const unsigned long long* __restrict__ best,
                           int* __restrict__ prev, float* __restrict__ out) {
  int b = threadIdx.x;                               // 64 threads
  int sampled = (int)(best[b] & 0xFFFFFFFFull);
  int tok = (t < *given) ? x[b * T_LEN + t] : sampled;
  prev[b] = tok;
  out[b * T_LEN + t] = (float)tok;
}

// ---------------- host ----------------

extern "C" void kernel_launch(void* const* d_in, const int* in_sizes, int n_in,
                              void* d_out, int out_size, void* d_ws,
                              size_t ws_size, hipStream_t stream) {
  const int*   x        = (const int*)d_in[0];
  const int*   cls_lab  = (const int*)d_in[1];
  const int*   given    = (const int*)d_in[2];
  const float* emb      = (const float*)d_in[3];
  const float* cls_tab  = (const float*)d_in[4];
  const float* W_ih     = (const float*)d_in[5];
  const float* W_hh     = (const float*)d_in[6];
  const float* b_ih     = (const float*)d_in[7];
  const float* b_hh     = (const float*)d_in[8];
  const float* W_out    = (const float*)d_in[9];
  const float* b_out    = (const float*)d_in[10];
  float*       out      = (float*)d_out;

  char* ws = (char*)d_ws;
  __hip_bfloat16* Wcat = (__hip_bfloat16*)ws; ws += (size_t)G4 * KCAT * 2;
  __hip_bfloat16* Wo   = (__hip_bfloat16*)ws; ws += (size_t)V_SZ * H_DIM * 2;
  __hip_bfloat16* xin  = (__hip_bfloat16*)ws; ws += (size_t)B_SZ * KCAT * 2;
  float* gates         = (float*)ws;          ws += (size_t)B_SZ * G4 * 4;
  float* cst           = (float*)ws;          ws += (size_t)B_SZ * H_DIM * 4;
  float* bcat          = (float*)ws;          ws += (size_t)G4 * 4;
  int* prev            = (int*)ws;            ws += 256;
  unsigned long long* best = (unsigned long long*)ws;

  // one-time (per call) weight conversion + state init
  k_cvt_wcat<<<(G4 * KCAT) / 256, 256, 0, stream>>>(W_ih, W_hh, Wcat);
  k_cvt_wout<<<(V_SZ * H_DIM) / (4 * 256), 256, 0, stream>>>(W_out, Wo);
  k_setup<<<256, 256, 0, stream>>>(cls_tab, cls_lab, x, b_ih, b_hh, cst, xin,
                                   bcat, prev);

  for (int t = 0; t < T_LEN; ++t) {
    k_build_input<<<B_SZ, E_DIM, 0, stream>>>(t, x, given, prev, emb, xin);
    k_gates<<<128, 256, 0, stream>>>(xin, Wcat, gates);          // 1024 waves
    k_cell<<<256, 256, 0, stream>>>(gates, bcat, cst, xin, best);
    k_logits<<<NT_OUT / 8, 256, 0, stream>>>(t, xin, Wo, b_out, best);
    k_finalize<<<1, B_SZ, 0, stream>>>(t, x, given, best, prev, out);
  }
}
